// SSM_10402410791637
// MI455X (gfx1250) — compile-verified, matrix operand order checked by reading, not run on hardware
//
#include <hip/hip_runtime.h>
#include <hip/hip_bf16.h>
#include <math.h>

// ---- problem constants (match reference) ----
#define BB    2
#define SS    2048
#define DIMK  2048      // DIM (K of input GEMM)
#define DIN   8480      // 2*HID + 2*G*N + H
#define HIDN  4096
#define NH    32        // heads
#define HP    128       // P = HID/H
#define NS    128       // G*N state dim
#define CONVD 4352      // HID + 2*G*N
#define NC    8         // chunks = S/CHUNK
#define LCH   256       // CHUNK
#define KCV   4         // conv width
#define MROWS (BB*SS)   // 4096

typedef __attribute__((ext_vector_type(16))) __bf16 v16bf;
typedef __attribute__((ext_vector_type(8)))  float  v8f;
typedef __attribute__((ext_vector_type(4)))  int    v4i;

// Async global->LDS path (CDNA5); fall back to sync copies if the toolchain
// doesn't expose the builtins.
#if defined(__gfx1250__) && __has_builtin(__builtin_amdgcn_global_load_async_to_lds_b128) && __has_builtin(__builtin_amdgcn_s_wait_asynccnt)
#define HAVE_ASYNC_LDS 1
#else
#define HAVE_ASYNC_LDS 0
#endif

__device__ __forceinline__ void async_copy16(const __bf16* g, __bf16* l) {
#if HAVE_ASYNC_LDS
  __builtin_amdgcn_global_load_async_to_lds_b128(
      (__attribute__((address_space(1))) v4i*)g,
      (__attribute__((address_space(3))) v4i*)l, 0, 0);
#else
  *(uint4*)l = *(const uint4*)g;
#endif
}

// ---------------------------------------------------------------------------
// wave-level 16x16 WMMA tile helper (direct-from-global operands)
// ---------------------------------------------------------------------------
__device__ __forceinline__ v8f tile_bf16(const __bf16* __restrict__ A, int lda,
                                         const __bf16* __restrict__ Bm, int ldb,
                                         int m0, int n0, int K) {
  const int lane = threadIdx.x & 31, r = lane & 15, half = lane >> 4;
  v8f acc = {0.f,0.f,0.f,0.f,0.f,0.f,0.f,0.f};
  const __bf16* ap = A  + (long)(m0 + r) * lda + half * 16;
  const __bf16* bp = Bm + (long)(n0 + r) * ldb + half * 16;
  for (int k0 = 0; k0 < K; k0 += 32) {
    v16bf a, b;
#pragma unroll
    for (int i = 0; i < 16; ++i) { a[i] = ap[i]; b[i] = bp[i]; }
    acc = __builtin_amdgcn_wmma_f32_16x16x32_bf16(false, a, false, b,
                                                  (short)0, acc, false, false);
    ap += 32; bp += 32;
  }
  return acc;
}

__device__ __forceinline__ void store_tile(float* __restrict__ C, int ldc,
                                           int m0, int n0, v8f acc) {
  const int lane = threadIdx.x & 31, r = lane & 15, half = lane >> 4;
#pragma unroll
  for (int i = 0; i < 8; ++i)
    C[(long)(m0 + i + half * 8) * ldc + n0 + r] = acc[i];
}

// ---------------------------------------------------------------------------
// fp32 -> bf16 cast
// ---------------------------------------------------------------------------
__global__ void k_cast_bf16(const float* __restrict__ src, __bf16* __restrict__ dst, long n) {
  long i = (long)blockIdx.x * blockDim.x + threadIdx.x;
  long stride = (long)gridDim.x * blockDim.x;
  for (; i < n; i += stride) dst[i] = (__bf16)src[i];
}

// ---------------------------------------------------------------------------
// bf16 GEMM: C[M,N] f32 = A[M,K] * B[N,K]^T.
// Block tile 64(M) x 64(N); K in steps of 32. A/B strips (64x32 bf16, 4 KB)
// double-buffered in LDS, filled by async global->LDS b128 copies (one per
// thread per strip), drained by 8 waves each computing 16x32 (2 accumulators,
// A fragment reused). Requires M%64==0, N%16==0, K%32==0.
// ---------------------------------------------------------------------------
__global__ void k_gemm_bf16(const __bf16* __restrict__ A, const __bf16* __restrict__ Bm,
                            float* __restrict__ C, int M, int N, int K,
                            int lda, int ldb, int ldc) {
  __shared__ __attribute__((aligned(16))) __bf16 lA[2][64 * 32];
  __shared__ __attribute__((aligned(16))) __bf16 lB[2][64 * 32];
  const int tid = threadIdx.x;
  const int bm = blockIdx.y * 64;
  const int bn = blockIdx.x * 64;

  // staging map: 256 threads x 16 B cover one 64x32 bf16 strip
  const int srow = tid >> 2;
  const int sseg = (tid & 3) * 8;
  int brow = bn + srow; if (brow > N - 1) brow = N - 1;   // clamp ragged N
  const __bf16* gA = A  + (long)(bm + srow) * lda + sseg;
  const __bf16* gB = Bm + (long)brow * ldb + sseg;
  __bf16* sA = &lA[0][srow * 32 + sseg];
  __bf16* sB = &lB[0][srow * 32 + sseg];

  const int wid = tid >> 5, lane = tid & 31, r = lane & 15, half = lane >> 4;
  const int tm = (wid >> 1) * 16;
  const int tn = (wid & 1) * 32;
  v8f acc0 = {0.f,0.f,0.f,0.f,0.f,0.f,0.f,0.f};
  v8f acc1 = {0.f,0.f,0.f,0.f,0.f,0.f,0.f,0.f};

  const int KT = K >> 5;
  async_copy16(gA, sA);
  async_copy16(gB, sB);
  for (int kt = 0; kt < KT; ++kt) {
    const int cur = kt & 1;
    if (kt + 1 < KT) {
      const int nxt = cur ^ 1;
      async_copy16(gA + (long)(kt + 1) * 32, sA + nxt * (64 * 32));
      async_copy16(gB + (long)(kt + 1) * 32, sB + nxt * (64 * 32));
      __builtin_prefetch(gA + (long)(kt + 2) * 32, 0, 1);
      __builtin_prefetch(gB + (long)(kt + 2) * 32, 0, 1);
#if HAVE_ASYNC_LDS
      __builtin_amdgcn_s_wait_asynccnt(2);   // stage kt complete (per wave)
#endif
    } else {
#if HAVE_ASYNC_LDS
      __builtin_amdgcn_s_wait_asynccnt(0);
#endif
    }
    __syncthreads();                         // cross-wave visibility
    const __bf16* pa  = &lA[cur][(tm + r) * 32 + half * 16];
    const __bf16* pb0 = &lB[cur][(tn + r) * 32 + half * 16];
    const __bf16* pb1 = &lB[cur][(tn + 16 + r) * 32 + half * 16];
    v16bf a, b0, b1;
#pragma unroll
    for (int i = 0; i < 16; ++i) { a[i] = pa[i]; b0[i] = pb0[i]; b1[i] = pb1[i]; }
    acc0 = __builtin_amdgcn_wmma_f32_16x16x32_bf16(false, a, false, b0,
                                                   (short)0, acc0, false, false);
    acc1 = __builtin_amdgcn_wmma_f32_16x16x32_bf16(false, a, false, b1,
                                                   (short)0, acc1, false, false);
    __syncthreads();                         // buffer reuse safety
  }
  const int m0 = bm + tm;
  const int n0 = bn + tn;
  if (n0 < N) {
#pragma unroll
    for (int i = 0; i < 8; ++i)
      C[(long)(m0 + i + half * 8) * ldc + n0 + r] = acc0[i];
  }
  if (n0 + 16 < N) {
#pragma unroll
    for (int i = 0; i < 8; ++i)
      C[(long)(m0 + i + half * 8) * ldc + n0 + 16 + r] = acc1[i];
  }
}

// ---------------------------------------------------------------------------
// causal conv(K=4) + SiLU over xBC; split into xs / B / C, and emit the
// WMMA-friendly transposed bf16 layouts: xsT[b,h,p,s], B^T[b,n,s].
// ---------------------------------------------------------------------------
__global__ void k_conv_silu(const float* __restrict__ zx, const float* __restrict__ cw,
                            const float* __restrict__ cb,
                            float* __restrict__ xs, float* __restrict__ Bf,
                            float* __restrict__ Cf,
                            __bf16* __restrict__ Bbf, __bf16* __restrict__ Cbf,
                            __bf16* __restrict__ xsTb, __bf16* __restrict__ BbfT) {
  const long total = (long)BB * SS * CONVD;
  long idx = (long)blockIdx.x * blockDim.x + threadIdx.x;
  const long stride = (long)gridDim.x * blockDim.x;
  for (; idx < total; idx += stride) {
    int ch = (int)(idx % CONVD);
    long bt = idx / CONVD;
    int t = (int)(bt % SS);
    int b = (int)(bt / SS);
    float acc = cb[ch];
#pragma unroll
    for (int k = 0; k < KCV; ++k) {
      int ti = t - (KCV - 1) + k;
      if (ti >= 0)
        acc += zx[((long)b * SS + ti) * DIN + HIDN + ch] * cw[ch * KCV + k];
    }
    float v = acc / (1.f + expf(-acc));   // SiLU
    long row = (long)b * SS + t;
    if (ch < HIDN) {
      xs[row * HIDN + ch] = v;
      int h = ch / HP, p = ch % HP;
      xsTb[((long)(b * NH + h) * HP + p) * SS + t] = (__bf16)v;
    } else if (ch < HIDN + NS) {
      int n = ch - HIDN;
      Bf[row * NS + n] = v;
      Bbf[row * NS + n] = (__bf16)v;
      BbfT[((long)b * NS + n) * SS + t] = (__bf16)v;
    } else {
      int n = ch - HIDN - NS;
      Cf[row * NS + n] = v;
      Cbf[row * NS + n] = (__bf16)v;
    }
  }
}

// ---------------------------------------------------------------------------
// dt = clip(softplus(dt_raw + dt_bias), 1e-3, 0.1)
// ---------------------------------------------------------------------------
__global__ void k_dt(const float* __restrict__ zx, const float* __restrict__ dt_bias,
                     float* __restrict__ dt) {
  int idx = blockIdx.x * blockDim.x + threadIdx.x;
  if (idx >= BB * SS * NH) return;
  int h = idx % NH;
  long row = idx / NH;
  float raw = zx[row * DIN + (2 * HIDN + 2 * NS) + h] + dt_bias[h];
  float sp = (raw > 20.f) ? raw : log1pf(expf(raw));
  sp = fminf(fmaxf(sp, 1e-3f), 0.1f);
  dt[row * NH + h] = sp;
}

// ---------------------------------------------------------------------------
// per (b, chunk, head) cumulative sum of dA = dt * A over L
// ---------------------------------------------------------------------------
__global__ void k_cumsum(const float* __restrict__ dt, const float* __restrict__ A_log,
                         float* __restrict__ dAcs) {
  int idx = blockIdx.x * blockDim.x + threadIdx.x;
  if (idx >= BB * NC * NH) return;
  int h = idx % NH;
  int bc = idx / NH;                 // b*NC + c
  float A = -expf(A_log[h]);
  float run = 0.f;
  for (int l = 0; l < LCH; ++l) {
    run += dt[((long)bc * LCH + l) * NH + h] * A;
    dAcs[((long)bc * LCH + l) * NH + h] = run;
  }
}

// ---------------------------------------------------------------------------
// CB[b,c] = Cc(LxN) @ Bc(LxN)^T   (per-chunk 256x256, K=128) via WMMA
// ---------------------------------------------------------------------------
__global__ void k_cb(const __bf16* __restrict__ Cbf, const __bf16* __restrict__ Bbf,
                     float* __restrict__ CBuf) {
  const int bc = blockIdx.z;
  const __bf16* Ab = Cbf + (long)bc * LCH * NS;
  const __bf16* Bb = Bbf + (long)bc * LCH * NS;
  const int wid = threadIdx.x >> 5, wy = wid >> 2, wx = wid & 3;
  const int m0 = blockIdx.y * 32 + wy * 16;
  const int n0 = blockIdx.x * 64 + wx * 16;
  v8f acc = tile_bf16(Ab, NS, Bb, NS, m0, n0, NS);
  store_tile(CBuf + (long)bc * LCH * LCH, LCH, m0, n0, acc);
}

// ---------------------------------------------------------------------------
// chunk states: states[b,c,h,p,n] = sum_l exp(dAcs[L-1]-dAcs[l])*dt[l]*x[l,p]*B[l,n]
// GEMM (P=128) x (K=L=256) x (N=128) per (b,c,h). A/B operands read from
// s-contiguous transposed bf16 layouts.
// ---------------------------------------------------------------------------
__global__ void k_states(const __bf16* __restrict__ xsTb, const __bf16* __restrict__ BbfT,
                         const float* __restrict__ dt, const float* __restrict__ dAcs,
                         float* __restrict__ states) {
  const int bch = blockIdx.x;        // (b*NC+c)*NH + h
  const int h = bch % NH;
  const int bc = bch / NH;
  const int b = bc / NC, c = bc % NC;
  __shared__ float w[LCH];
  const int tid = threadIdx.x;
  {
    int l = tid;
    float dal   = dAcs[((long)bc * LCH + l) * NH + h];
    float dalst = dAcs[((long)bc * LCH + (LCH - 1)) * NH + h];
    w[l] = expf(dalst - dal) * dt[((long)bc * LCH + l) * NH + h];
  }
  __syncthreads();
  const int wid = tid >> 5, lane = tid & 31, r = lane & 15, half = lane >> 4;
  const int p0 = wid * 16;
  const __bf16* xtb = xsTb + ((long)(b * NH + h) * HP) * SS + c * LCH;  // [p][s]
  const __bf16* btb = BbfT + ((long)b * NS) * SS + c * LCH;             // [n][s]
  float* stout = states + (long)bch * HP * NS;
  for (int nt = 0; nt < 8; ++nt) {
    const int n0 = nt * 16;
    v8f acc = {0.f,0.f,0.f,0.f,0.f,0.f,0.f,0.f};
    for (int k0 = 0; k0 < LCH; k0 += 32) {
      const int kk = k0 + half * 16;
      v16bf a, bfr;
#pragma unroll
      for (int i = 0; i < 16; ++i) {
        int l = kk + i;
        a[i]   = (__bf16)((float)xtb[(long)(p0 + r) * SS + l] * w[l]); // A[p,l]
        bfr[i] = btb[(long)(n0 + r) * SS + l];                         // Bop[n,l]
      }
      acc = __builtin_amdgcn_wmma_f32_16x16x32_bf16(false, a, false, bfr,
                                                    (short)0, acc, false, false);
    }
#pragma unroll
    for (int i = 0; i < 8; ++i)
      stout[(long)(p0 + i + half * 8) * NS + n0 + r] = acc[i];
  }
}

// ---------------------------------------------------------------------------
// inter-chunk state recurrence (sequential over NC=8 chunks per (b,h))
// ---------------------------------------------------------------------------
__global__ void k_scan(const float* __restrict__ states, const float* __restrict__ dAcs,
                       float* __restrict__ prevst) {
  const int bh = blockIdx.x;          // b*NH + h
  const int b = bh / NH, h = bh % NH;
  const int tid = threadIdx.x;
  float carry[64];
#pragma unroll
  for (int j = 0; j < 64; ++j) carry[j] = 0.f;
  for (int c = 0; c < NC; ++c) {
    const long sb = ((long)(b * NC + c) * NH + h) * (long)HP * NS;
    const float dec = expf(dAcs[((long)(b * NC + c) * LCH + (LCH - 1)) * NH + h]);
    for (int j = 0; j < 64; ++j) {
      long e = sb + (long)j * 256 + tid;
      prevst[e] = carry[j];
      carry[j] = carry[j] * dec + states[e];
    }
  }
}

// ---------------------------------------------------------------------------
// y = y_intra + y_inter + D*x   per (b,c,h).
// intra: M_h[l,s] staged in LDS (bf16, 16-row strips), y_intra = M_h @ X.
// inter: y_inter = (C*exp(dAcs)) @ prev_state^T.
// ---------------------------------------------------------------------------
__global__ void k_y(const float* __restrict__ xs, const __bf16* __restrict__ xsTb,
                    const float* __restrict__ Cf,
                    const float* __restrict__ dt, const float* __restrict__ dAcs,
                    const float* __restrict__ CBuf, const float* __restrict__ prevst,
                    const float* __restrict__ Dp, float* __restrict__ y) {
  const int bch = blockIdx.x;
  const int h = bch % NH;
  const int bc = bch / NH;
  const int b = bc / NC, c = bc % NC;
  __shared__ float s_dt[LCH], s_dA[LCH];
  __shared__ __attribute__((aligned(16))) __bf16 Msh[16 * LCH];
  const int tid = threadIdx.x;
  s_dt[tid] = dt[((long)bc * LCH + tid) * NH + h];
  s_dA[tid] = dAcs[((long)bc * LCH + tid) * NH + h];
  __syncthreads();
  const int wid = tid >> 5, lane = tid & 31, r = lane & 15, half = lane >> 4;
  const int p0 = wid * 16;
  const float* xsb  = xs + (long)bc * LCH * HIDN + h * HP;
  const __bf16* xtb = xsTb + ((long)(b * NH + h) * HP) * SS + c * LCH;  // [p][s]
  const float* Cb   = Cf + (long)bc * LCH * NS;
  const float* CBb  = CBuf + (long)bc * LCH * LCH;
  const float* pst  = prevst + (long)bch * (long)HP * NS;
  float* yb = y + (long)bc * LCH * HIDN + h * HP;
  const float Dh = Dp[h];

  for (int lt = 0; lt < 16; ++lt) {
    const int l0 = lt * 16;
    {   // build 16x256 masked-decay strip of M_h in LDS (bf16)
      const int s = tid;
      const float dts = s_dt[s], dAs = s_dA[s];
#pragma unroll
      for (int i = 0; i < 16; ++i) {
        int l = l0 + i;
        float v = 0.f;
        if (s <= l) v = CBb[(long)l * LCH + s] * expf(s_dA[l] - dAs) * dts;
        Msh[i * LCH + s] = (__bf16)v;
      }
    }
    __syncthreads();

    v8f acc = {0.f,0.f,0.f,0.f,0.f,0.f,0.f,0.f};
    // intra-chunk: K = LCH over s (A from LDS, B from transposed bf16 x)
    for (int k0 = 0; k0 < LCH; k0 += 32) {
      const int kk = k0 + half * 16;
      v16bf a, bfr;
#pragma unroll
      for (int i = 0; i < 16; ++i) {
        a[i]   = Msh[r * LCH + kk + i];
        bfr[i] = xtb[(long)(p0 + r) * SS + kk + i];   // Bop[p,s] = x[s,p]
      }
      acc = __builtin_amdgcn_wmma_f32_16x16x32_bf16(false, a, false, bfr,
                                                    (short)0, acc, false, false);
    }
    // inter-chunk: K = NS over n
    const float el = expf(s_dA[l0 + r]);
    for (int k0 = 0; k0 < NS; k0 += 32) {
      const int kk = k0 + half * 16;
      v16bf a, bfr;
#pragma unroll
      for (int i = 0; i < 16; ++i) {
        a[i]   = (__bf16)(Cb[(long)(l0 + r) * NS + kk + i] * el);
        bfr[i] = (__bf16)pst[(long)(p0 + r) * NS + kk + i];
      }
      acc = __builtin_amdgcn_wmma_f32_16x16x32_bf16(false, a, false, bfr,
                                                    (short)0, acc, false, false);
    }
    // store y tile (+ D * x skip term)
#pragma unroll
    for (int i = 0; i < 8; ++i) {
      int l = l0 + i + half * 8;
      float xv = xsb[(long)l * HIDN + p0 + r];
      yb[(long)l * HIDN + p0 + r] = acc[i] + Dh * xv;
    }
    __syncthreads();
  }
}

// ---------------------------------------------------------------------------
// y *= silu(z) ; RMSNorm(HID) * norm_w ; cast to bf16 for the output GEMM
// ---------------------------------------------------------------------------
__global__ void k_gate_norm(float* __restrict__ y, const float* __restrict__ zx,
                            const float* __restrict__ norm_w, __bf16* __restrict__ ybf) {
  const long row = blockIdx.x;
  const int tid = threadIdx.x;
  float ss = 0.f;
  for (int j = tid; j < HIDN; j += 256) {
    float z = zx[row * DIN + j];
    float g = y[row * HIDN + j] * (z / (1.f + expf(-z)));
    y[row * HIDN + j] = g;
    ss += g * g;
  }
  __shared__ float red[256];
  red[tid] = ss; __syncthreads();
  for (int off = 128; off > 0; off >>= 1) {
    if (tid < off) red[tid] += red[tid + off];
    __syncthreads();
  }
  const float scale = rsqrtf(red[0] / HIDN + 1e-6f);
  for (int j = tid; j < HIDN; j += 256)
    ybf[row * HIDN + j] = (__bf16)(y[row * HIDN + j] * scale * norm_w[j]);
}

// ---------------------------------------------------------------------------
// host launcher
// ---------------------------------------------------------------------------
extern "C" void kernel_launch(void* const* d_in, const int* in_sizes, int n_in,
                              void* d_out, int out_size, void* d_ws, size_t ws_size,
                              hipStream_t stream) {
  const float* x       = (const float*)d_in[0];
  const float* W_in    = (const float*)d_in[1];
  const float* conv_w  = (const float*)d_in[2];
  const float* conv_b  = (const float*)d_in[3];
  const float* dt_bias = (const float*)d_in[4];
  const float* A_log   = (const float*)d_in[5];
  const float* Dp      = (const float*)d_in[6];
  const float* norm_w  = (const float*)d_in[7];
  const float* W_out   = (const float*)d_in[8];
  float* out = (float*)d_out;

  char* base = (char*)d_ws;
  size_t off = 0;
  auto alloc = [&](size_t bytes) -> void* {
    off = (off + 255) & ~(size_t)255;
    void* p = base + off;
    off += bytes;
    return p;
  };

  __bf16* xbf    = (__bf16*)alloc((size_t)MROWS * DIMK * 2);
  __bf16* winbf  = (__bf16*)alloc((size_t)DIN * DIMK * 2);
  __bf16* woutbf = (__bf16*)alloc((size_t)DIMK * HIDN * 2);
  float*  zx     = (float*)alloc((size_t)MROWS * DIN * 4);
  float*  xsf    = (float*)alloc((size_t)MROWS * HIDN * 4);
  float*  Bf     = (float*)alloc((size_t)MROWS * NS * 4);
  float*  Cf     = (float*)alloc((size_t)MROWS * NS * 4);
  __bf16* Bbf    = (__bf16*)alloc((size_t)MROWS * NS * 2);
  __bf16* Cbf    = (__bf16*)alloc((size_t)MROWS * NS * 2);
  __bf16* xsTb   = (__bf16*)alloc((size_t)MROWS * HIDN * 2);
  __bf16* BbfT   = (__bf16*)alloc((size_t)BB * NS * SS * 2);
  float*  dtb    = (float*)alloc((size_t)MROWS * NH * 4);
  float*  dAcs   = (float*)alloc((size_t)BB * NC * LCH * NH * 4);
  float*  CBuf   = (float*)alloc((size_t)BB * NC * LCH * LCH * 4);
  float*  states = (float*)alloc((size_t)BB * NC * NH * HP * NS * 4);
  float*  prevst = (float*)alloc((size_t)BB * NC * NH * HP * NS * 4);
  float*  yf     = (float*)alloc((size_t)MROWS * HIDN * 4);
  __bf16* ybf    = (__bf16*)alloc((size_t)MROWS * HIDN * 2);

  dim3 blk(256);

  // bf16 casts of GEMM operands
  k_cast_bf16<<<4096, blk, 0, stream>>>(x,     xbf,    (long)MROWS * DIMK);
  k_cast_bf16<<<8192, blk, 0, stream>>>(W_in,  winbf,  (long)DIN * DIMK);
  k_cast_bf16<<<4096, blk, 0, stream>>>(W_out, woutbf, (long)DIMK * HIDN);

  // input projection: zx[4096, 8480] = x @ W_in^T
  k_gemm_bf16<<<dim3((DIN + 63) / 64, MROWS / 64), blk, 0, stream>>>(
      xbf, winbf, zx, MROWS, DIN, DIMK, DIMK, DIMK, DIN);

  // conv + SiLU + dt (+ transposed bf16 layouts)
  k_conv_silu<<<69632, blk, 0, stream>>>(zx, conv_w, conv_b, xsf, Bf, Cf,
                                         Bbf, Cbf, xsTb, BbfT);
  k_dt<<<(BB * SS * NH + 255) / 256, blk, 0, stream>>>(zx, dt_bias, dtb);

  // per-chunk cumsum of dA
  k_cumsum<<<(BB * NC * NH + 255) / 256, blk, 0, stream>>>(dtb, A_log, dAcs);

  // CB = Cc @ Bc^T per chunk
  k_cb<<<dim3(LCH / 64, LCH / 32, BB * NC), blk, 0, stream>>>(Cbf, Bbf, CBuf);

  // chunk states GEMM per (b,c,h)
  k_states<<<BB * NC * NH, blk, 0, stream>>>(xsTb, BbfT, dtb, dAcs, states);

  // inter-chunk recurrence per (b,h)
  k_scan<<<BB * NH, blk, 0, stream>>>(states, dAcs, prevst);

  // y = intra + inter + D*x per (b,c,h)
  k_y<<<BB * NC * NH, blk, 0, stream>>>(xsf, xsTb, Cf, dtb, dAcs, CBuf,
                                        prevst, Dp, yf);

  // gate + RMSNorm + cast
  k_gate_norm<<<MROWS, blk, 0, stream>>>(yf, zx, norm_w, ybf);

  // output projection: out[4096, 2048] = y_n @ W_out^T
  k_gemm_bf16<<<dim3(DIMK / 64, MROWS / 64), blk, 0, stream>>>(
      ybf, woutbf, out, MROWS, DIMK, HIDN, HIDN, HIDN, DIMK);
}
